// AttnDecoder2_33088428048483
// MI455X (gfx1250) — matrix operand order repaired and checked by reading, not compiled, not run on hardware
//
#include <hip/hip_runtime.h>

// ---------------------------------------------------------------------------
// AttnDecoder on MI455X (gfx1250, wave32):
//  - all weights converted fp32->bf16 once per launch (56 MB, L2-resident)
//  - every matmul runs on v_wmma_f32_16x16x32_bf16, f32 accumulation
//  - concat inputs handled by two sequential K-passes (no per-iter selects)
//  - inner loop software-pipelined: next A/B fragments load while current
//    4 WMMAs issue (kills WMMA->VALU WAR hazard NOPs)
//  - per-wave tile: 16 rows x 64 cols (4 accumulators, A-fragment reuse)
// ---------------------------------------------------------------------------

typedef __attribute__((ext_vector_type(16))) __bf16 v16bf;
typedef __attribute__((ext_vector_type(8)))  float  v8f;
typedef __attribute__((ext_vector_type(4)))  unsigned int uint4v;

union BfTile { uint4v q[2]; v16bf v; };

constexpr int BB = 64;    // batch
constexpr int LL = 96;    // sequence length
constexpr int HH = 1024;  // hidden
constexpr int MM = 512;   // memory (attention) length

__device__ __forceinline__ unsigned short f2bf(float x) {
  unsigned int u = __float_as_uint(x);
  u += 0x7FFFu + ((u >> 16) & 1u);   // round-to-nearest-even
  return (unsigned short)(u >> 16);
}

__device__ __forceinline__ void ld_a(BfTile& t, const unsigned short* p) {
  t.q[0] = *(const uint4v*)(p);
  t.q[1] = *(const uint4v*)(p + 8);
}
__device__ __forceinline__ void ld_b(BfTile& t, const unsigned short* p) {
  t.q[0] = *(const uint4v*)(p);
  t.q[1] = *(const uint4v*)(p + 8);
}

// One K-pass of the GEMM: acc += A(16 x K) @ W(64cols x K)^T for this wave's
// 16x64 tile. A row stride = lda, W row stride = ldw. K multiple of 32.
// Software-pipelined double buffer: next chunk's loads issue before current
// chunk's WMMAs so loads co-execute with matrix ops.
__device__ __forceinline__ void kpass(const unsigned short* __restrict__ A, int lda,
                                      const unsigned short* __restrict__ W, int ldw,
                                      int K, int m, int ncol0, int sel,
                                      v8f acc[4])
{
  const unsigned short* arow = A + (size_t)m * lda + (sel ? 16 : 0);
  const unsigned short* w0 = W + (size_t)(ncol0 +  0) * ldw + sel * 16;
  const unsigned short* w1 = W + (size_t)(ncol0 + 16) * ldw + sel * 16;
  const unsigned short* w2 = W + (size_t)(ncol0 + 32) * ldw + sel * 16;
  const unsigned short* w3 = W + (size_t)(ncol0 + 48) * ldw + sel * 16;

  BfTile a, b0, b1, b2, b3;
  ld_a(a, arow);
  ld_b(b0, w0); ld_b(b1, w1); ld_b(b2, w2); ld_b(b3, w3);

  for (int k = 0; k + 32 < K; k += 32) {
    arow += 32; w0 += 32; w1 += 32; w2 += 32; w3 += 32;
    BfTile an, bn0, bn1, bn2, bn3;
    ld_a(an, arow);
    ld_b(bn0, w0); ld_b(bn1, w1); ld_b(bn2, w2); ld_b(bn3, w3);

    acc[0] = __builtin_amdgcn_wmma_f32_16x16x32_bf16(
        false, a.v, false, b0.v, (short)0, acc[0], false, false);
    acc[1] = __builtin_amdgcn_wmma_f32_16x16x32_bf16(
        false, a.v, false, b1.v, (short)0, acc[1], false, false);
    acc[2] = __builtin_amdgcn_wmma_f32_16x16x32_bf16(
        false, a.v, false, b2.v, (short)0, acc[2], false, false);
    acc[3] = __builtin_amdgcn_wmma_f32_16x16x32_bf16(
        false, a.v, false, b3.v, (short)0, acc[3], false, false);

    a = an; b0 = bn0; b1 = bn1; b2 = bn2; b3 = bn3;
  }

  acc[0] = __builtin_amdgcn_wmma_f32_16x16x32_bf16(
      false, a.v, false, b0.v, (short)0, acc[0], false, false);
  acc[1] = __builtin_amdgcn_wmma_f32_16x16x32_bf16(
      false, a.v, false, b1.v, (short)0, acc[1], false, false);
  acc[2] = __builtin_amdgcn_wmma_f32_16x16x32_bf16(
      false, a.v, false, b2.v, (short)0, acc[2], false, false);
  acc[3] = __builtin_amdgcn_wmma_f32_16x16x32_bf16(
      false, a.v, false, b3.v, (short)0, acc[3], false, false);
}

// ---------------------------------------------------------------------------
// Generic bf16 WMMA GEMM:  Y(64 x N) = [A0 | A1](64 x (K0+K1)) @ [W0 | W1]^T
// W0 is N x K0 (row stride ldw0), W1 is N x K1 (row stride ldw1), both bf16.
// Optional dual bias (b0+b1), act: 0 = none, 1 = ReLU.
// Optional fp32 output (row stride ldf) and bf16 output (row stride ldb).
// One wave -> 16x64 tile. Grid = N/128 blocks of 256 threads (8 waves).
// ---------------------------------------------------------------------------
__global__ __launch_bounds__(256)
void gemm_bf16_kernel(const unsigned short* __restrict__ A0,
                      const unsigned short* __restrict__ A1,
                      int K0, int K1,
                      const unsigned short* __restrict__ W0, int ldw0,
                      const unsigned short* __restrict__ W1, int ldw1,
                      const float* __restrict__ bias0,
                      const float* __restrict__ bias1,
                      float* __restrict__ outF, int ldf,
                      unsigned short* __restrict__ outB, int ldb,
                      int N, int act)
{
  const int lane  = threadIdx.x & 31;
  const int gwave = blockIdx.x * 8 + (threadIdx.x >> 5);
  const int mtile = gwave & 3;      // 4 row tiles cover batch of 64
  const int ntile = gwave >> 2;     // each wave covers 64 output columns
  const int lh    = lane & 15;
  const int sel   = lane >> 4;      // half-wave select

  v8f acc[4];
#pragma unroll
  for (int nt = 0; nt < 4; ++nt) acc[nt] = (v8f)0.0f;

  const int m     = mtile * 16 + lh;
  const int ncol0 = ntile * 64 + lh;

  kpass(A0, K0, W0, ldw0, K0, m, ncol0, sel, acc);
  if (K1 > 0)
    kpass(A1, K1, W1, ldw1, K1, m, ncol0, sel, acc);

  // Epilogue. C/D layout: lane<16 -> rows 0..7 (VGPR r), lane>=16 -> rows 8..15.
#pragma unroll
  for (int nt = 0; nt < 4; ++nt) {
    const int n = ntile * 64 + nt * 16 + lh;
    float bsum = 0.0f;
    if (bias0) bsum += bias0[n];
    if (bias1) bsum += bias1[n];
#pragma unroll
    for (int r = 0; r < 8; ++r) {
      const int mr = mtile * 16 + sel * 8 + r;
      float v = acc[nt][r] + bsum;
      if (act == 1) v = fmaxf(v, 0.0f);
      if (outF) outF[(size_t)mr * ldf + n] = v;
      if (outB) outB[(size_t)mr * ldb + n] = f2bf(v);
    }
  }
}

// fp32 -> bf16 bulk conversion (weights, once per launch)
__global__ void f2bf_kernel(const float* __restrict__ src,
                            unsigned short* __restrict__ dst, int n)
{
  int i = blockIdx.x * 256 + threadIdx.x;
  if (i < n) dst[i] = f2bf(src[i]);
}

// Initialize recurrent state: inp = decode_input[:,0], h = h0 (bf16), c = c0
__global__ void init_state_kernel(const float* __restrict__ dec,
                                  const float* __restrict__ h0,
                                  const float* __restrict__ c0,
                                  unsigned short* __restrict__ inp_bf,
                                  unsigned short* __restrict__ h_bf,
                                  float* __restrict__ c_f)
{
  int idx = blockIdx.x * 256 + threadIdx.x;   // < BB*HH
  int b = idx / HH, hh = idx - b * HH;
  inp_bf[idx] = f2bf(dec[(size_t)b * LL * HH + hh]);
  h_bf[idx]   = f2bf(h0[idx]);
  c_f[idx]    = c0[idx];
}

// Row softmax over M=512; writes aw (fp32 workspace) and the attn output slice.
__global__ __launch_bounds__(256)
void softmax_kernel(const float* __restrict__ logits,
                    float* __restrict__ aw,
                    float* __restrict__ attn_out, int attn_ld)
{
  const int b = blockIdx.x;
  const int t = threadIdx.x;
  __shared__ float red[256];

  const float* row = logits + (size_t)b * MM;
  float v0 = row[t];
  float v1 = row[t + 256];

  red[t] = fmaxf(v0, v1);
  __syncthreads();
  for (int s = 128; s > 0; s >>= 1) {
    if (t < s) red[t] = fmaxf(red[t], red[t + s]);
    __syncthreads();
  }
  const float mx = red[0];
  __syncthreads();

  float e0 = __expf(v0 - mx);
  float e1 = __expf(v1 - mx);
  red[t] = e0 + e1;
  __syncthreads();
  for (int s = 128; s > 0; s >>= 1) {
    if (t < s) red[t] += red[t + s];
    __syncthreads();
  }
  const float inv = 1.0f / red[0];

  e0 *= inv; e1 *= inv;
  aw[(size_t)b * MM + t]        = e0;
  aw[(size_t)b * MM + t + 256]  = e1;
  attn_out[(size_t)b * attn_ld + t]       = e0;
  attn_out[(size_t)b * attn_ld + t + 256] = e1;
}

// applied[b,h] = sum_m aw[b,m] * enc[b,m,h]  (fp32 accumulate, bf16 out)
__global__ __launch_bounds__(256)
void attn_apply_kernel(const float* __restrict__ aw,
                       const float* __restrict__ enc,
                       unsigned short* __restrict__ applied_bf)
{
  const int b = blockIdx.y;
  const int h = blockIdx.x * 256 + threadIdx.x;
  __shared__ float s_aw[MM];
  for (int i = threadIdx.x; i < MM; i += 256) s_aw[i] = aw[(size_t)b * MM + i];
  __syncthreads();

  float acc = 0.0f;
  const float* e = enc + (size_t)b * MM * HH + h;
#pragma unroll 4
  for (int mm = 0; mm < MM; ++mm) acc += s_aw[mm] * e[(size_t)mm * HH];
  applied_bf[(size_t)b * HH + h] = f2bf(acc);
}

// LSTM cell elementwise: gates layout [i|f|g|o] along 4H; c updated in place,
// new h written as bf16 for the next GEMM.
__global__ __launch_bounds__(256)
void lstm_ew_kernel(const float* __restrict__ gates,
                    float* __restrict__ c,
                    unsigned short* __restrict__ h_bf)
{
  int idx = blockIdx.x * 256 + threadIdx.x;   // < BB*HH
  int b = idx / HH, hh = idx - b * HH;
  const float* g = gates + (size_t)b * 4 * HH;
  float gi = g[hh];
  float gf = g[HH + hh];
  float gg = g[2 * HH + hh];
  float go = g[3 * HH + hh];
  float si = 1.0f / (1.0f + __expf(-gi));
  float sf = 1.0f / (1.0f + __expf(-gf));
  float so = 1.0f / (1.0f + __expf(-go));
  float cn = sf * c[idx] + si * tanhf(gg);
  c[idx] = cn;
  h_bf[idx] = f2bf(so * tanhf(cn));
}

extern "C" void kernel_launch(void* const* d_in, const int* in_sizes, int n_in,
                              void* d_out, int out_size, void* d_ws, size_t ws_size,
                              hipStream_t stream) {
  (void)in_sizes; (void)n_in; (void)out_size; (void)ws_size;

  const float* decode_input = (const float*)d_in[0];
  const float* h0   = (const float*)d_in[1];
  const float* c0   = (const float*)d_in[2];
  const float* enc  = (const float*)d_in[3];
  const float* Wa   = (const float*)d_in[4];
  const float* ba   = (const float*)d_in[5];
  const float* Wc   = (const float*)d_in[6];
  const float* bc   = (const float*)d_in[7];
  const float* Wo   = (const float*)d_in[8];
  const float* bo   = (const float*)d_in[9];
  const float* W1ih = (const float*)d_in[10];
  const float* W1hh = (const float*)d_in[11];
  const float* b1ih = (const float*)d_in[12];
  const float* b1hh = (const float*)d_in[13];
  const float* W2ih = (const float*)d_in[14];
  const float* W2hh = (const float*)d_in[15];
  const float* b2ih = (const float*)d_in[16];
  const float* b2hh = (const float*)d_in[17];
  const float* W3ih = (const float*)d_in[18];
  const float* W3hh = (const float*)d_in[19];
  const float* b3ih = (const float*)d_in[20];
  const float* b3hh = (const float*)d_in[21];

  // ---- workspace carve-out (bf16 weights stay L2-resident all 96 steps) ----
  char* ws = (char*)d_ws;
  size_t off = 0;
  auto carve = [&](size_t bytes) -> void* {
    void* p = ws + off;
    off = (off + bytes + 255) & ~(size_t)255;
    return p;
  };
  unsigned short* Wa_bf   = (unsigned short*)carve((size_t)MM * 2 * HH * 2);
  unsigned short* Wc_bf   = (unsigned short*)carve((size_t)HH * 2 * HH * 2);
  unsigned short* Wo_bf   = (unsigned short*)carve((size_t)HH * HH * 2);
  unsigned short* W1ih_bf = (unsigned short*)carve((size_t)4 * HH * HH * 2);
  unsigned short* W1hh_bf = (unsigned short*)carve((size_t)4 * HH * HH * 2);
  unsigned short* W2ih_bf = (unsigned short*)carve((size_t)4 * HH * HH * 2);
  unsigned short* W2hh_bf = (unsigned short*)carve((size_t)4 * HH * HH * 2);
  unsigned short* W3ih_bf = (unsigned short*)carve((size_t)4 * HH * HH * 2);
  unsigned short* W3hh_bf = (unsigned short*)carve((size_t)4 * HH * HH * 2);
  unsigned short* inp_bf     = (unsigned short*)carve((size_t)BB * HH * 2);
  unsigned short* h_bf       = (unsigned short*)carve((size_t)BB * HH * 2);
  unsigned short* h1_bf      = (unsigned short*)carve((size_t)BB * HH * 2);
  unsigned short* h2_bf      = (unsigned short*)carve((size_t)BB * HH * 2);
  unsigned short* applied_bf = (unsigned short*)carve((size_t)BB * HH * 2);
  unsigned short* comb_bf    = (unsigned short*)carve((size_t)BB * HH * 2);
  float* c_f    = (float*)carve((size_t)BB * HH * 4);
  float* logits = (float*)carve((size_t)BB * MM * 4);
  float* aw     = (float*)carve((size_t)BB * MM * 4);
  float* gates  = (float*)carve((size_t)BB * 4 * HH * 4);

  auto conv = [&](const float* s, unsigned short* d, int n) {
    f2bf_kernel<<<(n + 255) / 256, 256, 0, stream>>>(s, d, n);
  };
  conv(Wa,   Wa_bf,   MM * 2 * HH);
  conv(Wc,   Wc_bf,   HH * 2 * HH);
  conv(Wo,   Wo_bf,   HH * HH);
  conv(W1ih, W1ih_bf, 4 * HH * HH);
  conv(W1hh, W1hh_bf, 4 * HH * HH);
  conv(W2ih, W2ih_bf, 4 * HH * HH);
  conv(W2hh, W2hh_bf, 4 * HH * HH);
  conv(W3ih, W3ih_bf, 4 * HH * HH);
  conv(W3hh, W3hh_bf, 4 * HH * HH);

  init_state_kernel<<<(BB * HH) / 256, 256, 0, stream>>>(
      decode_input, h0, c0, inp_bf, h_bf, c_f);

  auto gemm = [&](const unsigned short* A0, const unsigned short* A1, int K0, int K1,
                  const unsigned short* W0, int ldw0,
                  const unsigned short* W1, int ldw1,
                  const float* bias0, const float* bias1,
                  float* outF, int ldf, unsigned short* outB, int ldb,
                  int N, int act) {
    gemm_bf16_kernel<<<N / 128, 256, 0, stream>>>(
        A0, A1, K0, K1, W0, ldw0, W1, ldw1, bias0, bias1,
        outF, ldf, outB, ldb, N, act);
  };

  float* out_dec  = (float*)d_out;                        // B x L x H
  float* out_attn = (float*)d_out + (size_t)BB * LL * HH; // B x L x M

  for (int t = 0; t < LL; ++t) {
    // attention logits = [inp, h] @ Wa^T + ba
    gemm(inp_bf, h_bf, HH, HH, Wa_bf, 2 * HH, Wa_bf + HH, 2 * HH,
         ba, nullptr, logits, MM, nullptr, 0, MM, 0);
    // softmax -> aw (also writes attn output slice for step t)
    softmax_kernel<<<BB, 256, 0, stream>>>(
        logits, aw, out_attn + (size_t)t * MM, LL * MM);
    // applied = einsum('bm,bmh->bh', aw, enc)
    attn_apply_kernel<<<dim3(HH / 256, BB), 256, 0, stream>>>(
        aw, enc, applied_bf);
    // comb = relu([inp, applied] @ Wc^T + bc)
    gemm(inp_bf, applied_bf, HH, HH, Wc_bf, 2 * HH, Wc_bf + HH, 2 * HH,
         bc, nullptr, nullptr, 0, comb_bf, HH, HH, 1);
    // LSTM cell 1: gates = comb @ W1ih^T + h @ W1hh^T + (b1ih + b1hh)
    gemm(comb_bf, h_bf, HH, HH, W1ih_bf, HH, W1hh_bf, HH,
         b1ih, b1hh, gates, 4 * HH, nullptr, 0, 4 * HH, 0);
    lstm_ew_kernel<<<(BB * HH) / 256, 256, 0, stream>>>(gates, c_f, h1_bf);
    // LSTM cell 2 (x = h = h1)
    gemm(h1_bf, h1_bf, HH, HH, W2ih_bf, HH, W2hh_bf, HH,
         b2ih, b2hh, gates, 4 * HH, nullptr, 0, 4 * HH, 0);
    lstm_ew_kernel<<<(BB * HH) / 256, 256, 0, stream>>>(gates, c_f, h2_bf);
    // LSTM cell 3 (x = h = h2) -> new carried h
    gemm(h2_bf, h2_bf, HH, HH, W3ih_bf, HH, W3hh_bf, HH,
         b3ih, b3hh, gates, 4 * HH, nullptr, 0, 4 * HH, 0);
    lstm_ew_kernel<<<(BB * HH) / 256, 256, 0, stream>>>(gates, c_f, h_bf);
    // out = relu(h3 @ Wo^T + bo): fp32 to d_out, bf16 becomes next step's inp
    gemm(h_bf, nullptr, HH, 0, Wo_bf, HH, nullptr, 0,
         bo, nullptr, out_dec + (size_t)t * HH, LL * HH, inp_bf, HH, HH, 1);
  }
}
